// TreeModel_47021301957293
// MI455X (gfx1250) — matrix operand order repaired and checked by reading, not compiled, not run on hardware
//
#include <hip/hip_runtime.h>
#include <math.h>

// ---------------------------------------------------------------------------
// TreeModel on MI455X (gfx1250):
//   * one-time f32->bf16 conversion of x + all weights (weights pre-transposed
//     to [N][K] so every tile is K-contiguous)
//   * bf16 WMMA GEMMs with TDM (tensor_load_to_lds) double-buffered LDS staging
//   * wave32 reductions for the routing scalars
// B=16384, D_IN=256, E=512; all GEMM dims divide the 128x64 block tile.
// ---------------------------------------------------------------------------

#define BB   16384
#define DIN  256
#define EE   512
#define NIN  7
#define NLF  8

typedef __attribute__((ext_vector_type(16))) __bf16 v16bf;
typedef __attribute__((ext_vector_type(8)))  float  v8f;
typedef __attribute__((ext_vector_type(4)))  unsigned int tdm_v4u;
typedef __attribute__((ext_vector_type(8)))  int          tdm_v8i;
typedef __attribute__((ext_vector_type(4)))  int          tdm_v4i;

#if __has_builtin(__builtin_amdgcn_tensor_load_to_lds) && __has_builtin(__builtin_amdgcn_s_wait_tensorcnt)
#define HAVE_TDM 1
#else
#define HAVE_TDM 0
#endif

struct Frag { uint4 lo, hi; };   // 32 bytes = 16 bf16

static __device__ inline v16bf frag_as_bf16(const Frag& f) {
    v16bf r; __builtin_memcpy(&r, &f, 32); return r;
}

static __device__ inline unsigned short f2bf(float x) {
    unsigned u = __float_as_uint(x);
    unsigned r = (u + 0x7FFFu + ((u >> 16) & 1u)) >> 16;   // round-to-nearest-even
    return (unsigned short)r;
}
static __device__ inline float bf2f(unsigned short u) {
    return __uint_as_float((unsigned)u << 16);
}

static __device__ inline float wave_sum(float v) {
    #pragma unroll
    for (int o = 16; o > 0; o >>= 1) v += __shfl_xor(v, o, 32);
    return v;
}

// LDS byte offset: low 32 bits of the flat address of a __shared__ object
// (flat LDS aperture: addr[31:0] == LDS offset, ISA 10.2).
static __device__ inline unsigned lds_off(const void* p) {
    return (unsigned)(uintptr_t)p;
}

#if HAVE_TDM
// Issue a 2D TDM load: tile_h x tile_w bf16 elements, row stride = row_stride
// elements, packed contiguously into LDS at lds_byte. D# layout per ISA 8.3/8.4.
static __device__ inline void tdm_load_2d(unsigned lds_byte, const unsigned short* g,
                                          int row_stride, int tile_w, int tile_h,
                                          int tensor_w, int tensor_h) {
    unsigned long long ga = (unsigned long long)(uintptr_t)g;
    tdm_v4u g0;
    g0[0] = 1u;                                  // count=1, user descriptor
    g0[1] = lds_byte;                            // lds_addr (bytes)
    g0[2] = (unsigned)ga;                        // global_addr[31:0]
    g0[3] = (unsigned)(ga >> 32) | (2u << 30);   // global_addr[56:32] | type=2
    tdm_v8i g1;
    g1[0] = (int)(1u << 16);                     // wg_mask=0, data_size=1 (2B)
    g1[1] = (int)(((unsigned)tensor_w & 0xFFFFu) << 16);          // tensor_dim0 lo
    g1[2] = (int)((((unsigned)tensor_w >> 16) & 0xFFFFu) |
                  (((unsigned)tensor_h & 0xFFFFu) << 16));        // dim0 hi | dim1 lo
    g1[3] = (int)((((unsigned)tensor_h >> 16) & 0xFFFFu) |
                  ((unsigned)tile_w << 16));                      // dim1 hi | tile_dim0
    g1[4] = (int)(unsigned)tile_h;               // tile_dim1 | tile_dim2=0
    g1[5] = row_stride;                          // tensor_dim0_stride[31:0]
    g1[6] = 0;                                   // stride hi | tensor_dim1_stride lo
    g1[7] = 0;
    tdm_v4i gz4 = {0, 0, 0, 0};
    tdm_v8i gz8 = {0, 0, 0, 0, 0, 0, 0, 0};
    __builtin_amdgcn_tensor_load_to_lds(g0, g1, gz4, gz4, gz8, 0);
}
#endif

// ---------------------------------------------------------------------------
// C[M,N] = act(A[M,K] @ Wt[N,K]^T + bias[N]) ; ACT: 0 = none, 1 = relu
// A, Wt, C all bf16; bias f32. Block: 256 threads = 8 waves (4M x 2N), wave
// tile 32x32 (2x2 v_wmma_f32_16x16x32_bf16). LDS tiles double-buffered and
// filled by the Tensor Data Mover (wave 0 issues, s_wait_tensorcnt paces).
// ---------------------------------------------------------------------------
template <int ACT>
__global__ __launch_bounds__(256)
void gemm_bf16(const unsigned short* __restrict__ A,
               const unsigned short* __restrict__ Wt,
               const float* __restrict__ bias,
               unsigned short* __restrict__ C,
               int M, int N, int K) {
    const int bm   = blockIdx.x * 128;
    const int bn   = blockIdx.y * 64;
    const int tid  = threadIdx.x;
    const int lane = tid & 31;
    const int wave = tid >> 5;
    const int wm   = wave >> 1;   // 0..3 (M)
    const int wn   = wave & 1;    // 0..1 (N)

    __shared__ unsigned short lA[2][128 * 32];   // [m][k] tiles, 2 x 8 KB
    __shared__ unsigned short lB[2][64 * 32];    // [n][k] tiles, 2 x 4 KB

    v8f acc[2][2] = {};

    const int half = lane >> 4;
    const int l16  = lane & 15;
    const int nIter = K / 32;

#if HAVE_TDM
    if (wave == 0) {
        tdm_load_2d(lds_off(lA[0]), A  + (size_t)bm * K, K, 32, 128, K, M);
        tdm_load_2d(lds_off(lB[0]), Wt + (size_t)bn * K, K, 32, 64,  K, N);
    }
#endif

    for (int it = 0; it < nIter; ++it) {
        const int cur = it & 1;
#if HAVE_TDM
        if (wave == 0) {
            if (it + 1 < nIter) {   // prefetch next tile pair into other buffer
                const int nk = (it + 1) * 32;
                tdm_load_2d(lds_off(lA[cur ^ 1]), A  + (size_t)bm * K + nk, K, 32, 128, K, M);
                tdm_load_2d(lds_off(lB[cur ^ 1]), Wt + (size_t)bn * K + nk, K, 32, 64,  K, N);
                __builtin_amdgcn_s_wait_tensorcnt(2);   // pair 'it' done (in-order)
            } else {
                __builtin_amdgcn_s_wait_tensorcnt(0);
            }
        }
        __syncthreads();
#else
        __syncthreads();
        {   // cooperative bf16 copy fallback (no conversion needed)
            const int k0 = it * 32;
            const int m  = tid >> 1;                  // A: 128x32, 16 elems/thread
            const int kq = (tid & 1) * 16;
            const unsigned short* sa = A + (size_t)(bm + m) * K + k0 + kq;
            unsigned short* da = &lA[cur][m * 32 + kq];
            #pragma unroll
            for (int e = 0; e < 16; ++e) da[e] = sa[e];
            const int n   = tid >> 2;                 // B: 64x32, 8 elems/thread
            const int kq2 = (tid & 3) * 8;
            const unsigned short* sb = Wt + (size_t)(bn + n) * K + k0 + kq2;
            unsigned short* db = &lB[cur][n * 32 + kq2];
            #pragma unroll
            for (int e = 0; e < 8; ++e) db[e] = sb[e];
        }
        __syncthreads();
#endif
        // A fragments: lane<16 -> K 0..7 / 16..23 ; lane>=16 -> K 8..15 / 24..31
        Frag fa[2];
        #pragma unroll
        for (int tm = 0; tm < 2; ++tm) {
            const int m0 = (wm * 32 + tm * 16 + l16) * 32;
            fa[tm].lo = *(const uint4*)&lA[cur][m0 + half * 8];
            fa[tm].hi = *(const uint4*)&lA[cur][m0 + 16 + half * 8];
        }
        // B fragments: lane<16 -> K 0..7 / 16..23 ; lane>=16 -> K 8..15 / 24..31 (col = l16)
        Frag fb[2];
        #pragma unroll
        for (int tn = 0; tn < 2; ++tn) {
            const int n0 = (wn * 32 + tn * 16 + l16) * 32;
            fb[tn].lo = *(const uint4*)&lB[cur][n0 + half * 8];
            fb[tn].hi = *(const uint4*)&lB[cur][n0 + 16 + half * 8];
        }
        #pragma unroll
        for (int tm = 0; tm < 2; ++tm)
            #pragma unroll
            for (int tn = 0; tn < 2; ++tn)
                acc[tm][tn] = __builtin_amdgcn_wmma_f32_16x16x32_bf16(
                    false, frag_as_bf16(fa[tm]),
                    false, frag_as_bf16(fb[tn]),
                    (short)0, acc[tm][tn], false, false);
        __syncthreads();
    }

    // epilogue: VGPR r holds M = r + 8*(lane>=16), N = lane&15 (ISA 7.12.2)
    #pragma unroll
    for (int tm = 0; tm < 2; ++tm) {
        #pragma unroll
        for (int tn = 0; tn < 2; ++tn) {
            const int ncol = bn + wn * 32 + tn * 16 + l16;
            const float bv = bias[ncol];
            unsigned short* crow =
                C + (size_t)(bm + wm * 32 + tm * 16 + half * 8) * N + ncol;
            #pragma unroll
            for (int r = 0; r < 8; ++r) {
                float v = acc[tm][tn][r] + bv;
                if (ACT == 1) v = fmaxf(v, 0.0f);
                crow[(size_t)r * N] = f2bf(v);
            }
        }
    }
}

// ---------------------------------------------------------------------------
// One-time conversions.
// ---------------------------------------------------------------------------
__global__ __launch_bounds__(256)
void convert_f32_to_bf16(const float* __restrict__ src,
                         unsigned short* __restrict__ dst, int n) {
    const int i = blockIdx.x * 256 + threadIdx.x;
    if (i < n) dst[i] = f2bf(src[i]);
}

// W[K,N] f32 -> Wt[N,K] bf16 (K contiguous)
__global__ __launch_bounds__(256)
void transpose_f32_to_bf16(const float* __restrict__ W,
                           unsigned short* __restrict__ Wt, int K, int N) {
    const int idx = blockIdx.x * 256 + threadIdx.x;
    if (idx >= K * N) return;
    const int n = idx / K;
    const int k = idx - n * K;
    Wt[idx] = f2bf(W[(size_t)k * N + n]);
}

// ---------------------------------------------------------------------------
// Inner node: b = tanh(h.wb+bb); beta = h.wbeta+bbeta; idx = argmax z;
// pr = [max(z)>0] * [beta*(x[idx]+b) >= 0]   (max<=0 -> reference NaN -> 0)
// One wave per sample. h, z are bf16; x f32.
// ---------------------------------------------------------------------------
__global__ __launch_bounds__(256)
void node_reduce(const unsigned short* __restrict__ h, const unsigned short* __restrict__ z,
                 const float* __restrict__ x,
                 const float* __restrict__ wb, const float* __restrict__ wbeta,
                 const float* __restrict__ bb_all, const float* __restrict__ bbeta_all,
                 int node, float* __restrict__ pr) {
    const int s    = blockIdx.x * 8 + (threadIdx.x >> 5);
    const int lane = threadIdx.x & 31;

    const unsigned short* hr = h + (size_t)s * EE;
    float d1 = 0.f, d2 = 0.f;
    #pragma unroll 4
    for (int e = lane; e < EE; e += 32) {
        const float hv = bf2f(hr[e]);
        d1 += hv * wb[e];
        d2 += hv * wbeta[e];
    }
    d1 = wave_sum(d1);
    d2 = wave_sum(d2);

    const unsigned short* zr = z + (size_t)s * DIN;
    float mv = -INFINITY; int mi = 0x7fffffff;
    #pragma unroll 4
    for (int e = lane; e < DIN; e += 32) {
        const float v = bf2f(zr[e]);
        if (v > mv) { mv = v; mi = e; }
    }
    #pragma unroll
    for (int o = 16; o > 0; o >>= 1) {
        const float ov = __shfl_xor(mv, o, 32);
        const int   oi = __shfl_xor(mi, o, 32);
        if (ov > mv || (ov == mv && oi < mi)) { mv = ov; mi = oi; }
    }

    if (lane == 0) {
        const float b    = tanhf(d1 + bb_all[node]);
        const float beta = d2 + bbeta_all[node];
        float p = 0.f;
        if (mv > 0.f) {
            const float logit = beta * (x[(size_t)s * DIN + mi] + b);
            p = (logit >= 0.f) ? 1.f : 0.f;       // sigmoid(t)>=0.5 <=> t>=0
        }
        pr[s] = p;
    }
}

__global__ __launch_bounds__(256)
void leaf_reduce(const unsigned short* __restrict__ h, const float* __restrict__ w2,
                 const float* __restrict__ b2_all, int leaf,
                 float* __restrict__ leafout) {
    const int s    = blockIdx.x * 8 + (threadIdx.x >> 5);
    const int lane = threadIdx.x & 31;
    const unsigned short* hr = h + (size_t)s * EE;
    float d = 0.f;
    #pragma unroll 4
    for (int e = lane; e < EE; e += 32) d += bf2f(hr[e]) * w2[e];
    d = wave_sum(d);
    if (lane == 0) {
        const float t = d + b2_all[leaf];
        leafout[s] = 1.f / (1.f + expf(-t));
    }
}

__global__ __launch_bounds__(256)
void route_kernel(const float* __restrict__ pr, const float* __restrict__ leafout,
                  float* __restrict__ out) {
    const int s = blockIdx.x * 256 + threadIdx.x;
    if (s >= BB) return;
    int n = 0;
    #pragma unroll
    for (int d = 0; d < 3; ++d) {
        const float p = pr[(size_t)n * BB + s];
        n = 2 * n + 1 + (p >= 0.5f ? 1 : 0);
    }
    out[s] = leafout[(size_t)(n - NIN) * BB + s];
}

// ---------------------------------------------------------------------------
extern "C" void kernel_launch(void* const* d_in, const int* in_sizes, int n_in,
                              void* d_out, int out_size, void* d_ws, size_t ws_size,
                              hipStream_t stream) {
    const float* x       = (const float*)d_in[0];
    const float* enc_W0  = (const float*)d_in[1];
    const float* enc_b0  = (const float*)d_in[2];
    const float* enc_W   = (const float*)d_in[3];   // (3,E,E)
    const float* enc_b   = (const float*)d_in[4];   // (3,E)
    const float* in_W1   = (const float*)d_in[5];   // (7,E,E)
    const float* in_b1   = (const float*)d_in[6];   // (7,E)
    const float* in_W2   = (const float*)d_in[7];   // (7,E,DIN)
    const float* in_b2   = (const float*)d_in[8];   // (7,DIN)
    const float* in_wb   = (const float*)d_in[9];   // (7,E)
    const float* in_bb   = (const float*)d_in[10];  // (7,)
    const float* in_wbt  = (const float*)d_in[11];  // (7,E)
    const float* in_bbt  = (const float*)d_in[12];  // (7,)
    const float* lf_W1   = (const float*)d_in[13];  // (8,E,E)
    const float* lf_b1   = (const float*)d_in[14];  // (8,E)
    const float* lf_w2   = (const float*)d_in[15];  // (8,E)
    const float* lf_b2   = (const float*)d_in[16];  // (8,)
    float* out = (float*)d_out;

    // ---- workspace layout ----
    unsigned short* xbf  = (unsigned short*)d_ws;            // B*DIN
    unsigned short* actA = xbf  + (size_t)BB * DIN;          // B*E
    unsigned short* actB = actA + (size_t)BB * EE;           // B*E
    unsigned short* zbuf = actB + (size_t)BB * EE;           // B*DIN
    unsigned short* wt   = zbuf + (size_t)BB * DIN;          // weight pool
    unsigned short* encW0t = wt;                             // E x DIN
    unsigned short* encWt  = encW0t + (size_t)EE * DIN;      // 3 x E x E
    unsigned short* inW1t  = encWt  + (size_t)3 * EE * EE;   // 7 x E x E
    unsigned short* inW2t  = inW1t  + (size_t)NIN * EE * EE; // 7 x DIN x E
    unsigned short* lfW1t  = inW2t  + (size_t)NIN * DIN * EE;// 8 x E x E
    unsigned short* wend   = lfW1t  + (size_t)NLF * EE * EE;
    float* prb  = (float*)wend;                              // 7*B
    float* lout = prb + (size_t)NIN * BB;                    // 8*B

    const dim3 blk(256);
    const dim3 gE(BB / 128, EE / 64);
    const dim3 gD(BB / 128, DIN / 64);
    const dim3 gR(BB / 8);
    auto cgrid = [](size_t n) { return dim3((unsigned)((n + 255) / 256)); };

    // ---- one-time bf16 conversion / weight transposition ----
    convert_f32_to_bf16<<<cgrid((size_t)BB * DIN), blk, 0, stream>>>(x, xbf, BB * DIN);
    transpose_f32_to_bf16<<<cgrid((size_t)DIN * EE), blk, 0, stream>>>(enc_W0, encW0t, DIN, EE);
    for (int i = 0; i < 3; ++i)
        transpose_f32_to_bf16<<<cgrid((size_t)EE * EE), blk, 0, stream>>>(
            enc_W + (size_t)i * EE * EE, encWt + (size_t)i * EE * EE, EE, EE);
    for (int i = 0; i < NIN; ++i) {
        transpose_f32_to_bf16<<<cgrid((size_t)EE * EE), blk, 0, stream>>>(
            in_W1 + (size_t)i * EE * EE, inW1t + (size_t)i * EE * EE, EE, EE);
        transpose_f32_to_bf16<<<cgrid((size_t)EE * DIN), blk, 0, stream>>>(
            in_W2 + (size_t)i * EE * DIN, inW2t + (size_t)i * DIN * EE, EE, DIN);
    }
    for (int li = 0; li < NLF; ++li)
        transpose_f32_to_bf16<<<cgrid((size_t)EE * EE), blk, 0, stream>>>(
            lf_W1 + (size_t)li * EE * EE, lfW1t + (size_t)li * EE * EE, EE, EE);

    // ---- encoder: xbf -> actA -> actB -> actA -> actB (= emb) ----
    gemm_bf16<1><<<gE, blk, 0, stream>>>(xbf,  encW0t,                    enc_b0,          actA, BB, EE, DIN);
    gemm_bf16<1><<<gE, blk, 0, stream>>>(actA, encWt + (size_t)0 * EE * EE, enc_b + 0 * EE, actB, BB, EE, EE);
    gemm_bf16<1><<<gE, blk, 0, stream>>>(actB, encWt + (size_t)1 * EE * EE, enc_b + 1 * EE, actA, BB, EE, EE);
    gemm_bf16<1><<<gE, blk, 0, stream>>>(actA, encWt + (size_t)2 * EE * EE, enc_b + 2 * EE, actB, BB, EE, EE);
    const unsigned short* emb = actB;
    unsigned short* hbuf = actA;

    // ---- inner nodes ----
    for (int i = 0; i < NIN; ++i) {
        gemm_bf16<1><<<gE, blk, 0, stream>>>(emb, inW1t + (size_t)i * EE * EE,
                                             in_b1 + (size_t)i * EE, hbuf, BB, EE, EE);
        gemm_bf16<0><<<gD, blk, 0, stream>>>(hbuf, inW2t + (size_t)i * DIN * EE,
                                             in_b2 + (size_t)i * DIN, zbuf, BB, DIN, EE);
        node_reduce<<<gR, blk, 0, stream>>>(hbuf, zbuf, x,
                                            in_wb + (size_t)i * EE, in_wbt + (size_t)i * EE,
                                            in_bb, in_bbt, i, prb + (size_t)i * BB);
    }

    // ---- leaves ----
    for (int li = 0; li < NLF; ++li) {
        gemm_bf16<1><<<gE, blk, 0, stream>>>(emb, lfW1t + (size_t)li * EE * EE,
                                             lf_b1 + (size_t)li * EE, hbuf, BB, EE, EE);
        leaf_reduce<<<gR, blk, 0, stream>>>(hbuf, lf_w2 + (size_t)li * EE, lf_b2, li,
                                            lout + (size_t)li * BB);
    }

    route_kernel<<<dim3(BB / 256), blk, 0, stream>>>(prb, lout, out);
}
